// NonLocalBlock3D_66056597013148
// MI455X (gfx1250) — matrix operand order repaired and checked by reading, not compiled, not run on hardware
//
#include <hip/hip_runtime.h>
#include <hip/hip_bf16.h>

typedef __attribute__((ext_vector_type(16))) _Float16 v16h;
typedef __attribute__((ext_vector_type(8)))  _Float16 v8h;
typedef __attribute__((ext_vector_type(8)))  float    v8f;
typedef __attribute__((ext_vector_type(4)))  unsigned int u32x4;
typedef __attribute__((ext_vector_type(8)))  int          i32x8;
typedef __attribute__((ext_vector_type(4)))  int          i32x4;

union HV {
    v16h v;
    v8h  h2[2];
    _Float16 h[16];
};

#define NLB_N  16384   // D*H*W = 16*32*32
#define NLB_C  256
#define NLB_CR 128
#define NLB_NT (NLB_N / 64)      // 256 key tiles
#define NLB_SCALE 0.0625f        // 1/sqrt(C) = 1/16

static __device__ __forceinline__ v8f wmma_f16(v16h a, v16h b, v8f c) {
    return __builtin_amdgcn_wmma_f32_16x16x32_f16(false, a, false, b, (short)0, c, false, false);
}

// Issue a TDM load of a 64-row x 128-col f16 tile (row-major, contiguous)
// from global `src` into LDS byte offset `lds_byte`.
// D# layout per CDNA5 ISA ch.8 (2-D tile, groups 2/3 zero => up-to-2D tensor).
static __device__ __forceinline__ void tdm_load_k_tile(const _Float16* src,
                                                       unsigned lds_byte) {
    unsigned long long ga = (unsigned long long)(const void*)src;
    u32x4 g0;
    g0[0] = 1u;                                    // count=1, user descriptor
    g0[1] = lds_byte;                              // lds_addr (bytes)
    g0[2] = (unsigned)ga;                          // global_addr[31:0]
    g0[3] = (unsigned)((ga >> 32) & 0x01FFFFFFu)   // global_addr[56:32]
          | (2u << 30);                            // type = 2 ("image")
    i32x8 g1;
    g1[0] = (int)(1u << 16);        // data_size = 1 -> 2 bytes; wg_mask = 0
    g1[1] = (int)(128u << 16);      // tensor_dim0[15:0] = 128 @ bits 63:48
    g1[2] = (int)(16384u << 16);    // tensor_dim1[15:0] = 16384 @ bits 95:80
    g1[3] = (int)(128u << 16);      // tile_dim0 = 128 @ bits 127:112
    g1[4] = 64;                     // tile_dim1 = 64 @ bits 143:128
    g1[5] = 128;                    // tensor_dim0_stride = 128 @ bits 207:160
    g1[6] = 0;
    g1[7] = 0;
    i32x4 z4 = {0, 0, 0, 0};
    i32x8 z8 = {0, 0, 0, 0, 0, 0, 0, 0};
    // clang-23 / therock-10.0 form: 6 args (g0, g1, g2, g3, g_extra, cpol)
    __builtin_amdgcn_tensor_load_to_lds(g0, g1, z4, z4, z8, 0);
}

// ---------------------------------------------------------------------------
// Stage 1: Q/K/V projections.  dst[n][o] (f16) = W[o][:] . x[:][n] + bias[o]
// grid = (N/16, 3), block = 256 (8 waves; wave w owns output-channel tile w)
// ---------------------------------------------------------------------------
__global__ __launch_bounds__(256) void nlb_qkv_kernel(
    const float* __restrict__ x,
    const float* __restrict__ Wq, const float* __restrict__ bq,
    const float* __restrict__ Wk, const float* __restrict__ bk,
    const float* __restrict__ Wv, const float* __restrict__ bv,
    _Float16* __restrict__ Qh, _Float16* __restrict__ Kh, _Float16* __restrict__ Vh)
{
    const float* W    = (blockIdx.y == 0) ? Wq : (blockIdx.y == 1) ? Wk : Wv;
    const float* bias = (blockIdx.y == 0) ? bq : (blockIdx.y == 1) ? bk : bv;
    _Float16*    dst  = (blockIdx.y == 0) ? Qh : (blockIdx.y == 1) ? Kh : Vh;

    __shared__ __attribute__((aligned(32))) float ldsXT[16 * 32]; // [n][k]

    const int n0   = blockIdx.x * 16;
    const int wave = threadIdx.x >> 5;
    const int lane = threadIdx.x & 31;
    const int g    = lane >> 4;
    const int r    = lane & 15;
    const int o0   = wave * 16;

    v8f acc = {};
    for (int c0 = 0; c0 < NLB_C; c0 += 32) {
        #pragma unroll
        for (int idx = threadIdx.x; idx < 512; idx += 256) {
            int k = idx >> 4, n = idx & 15;
            ldsXT[n * 32 + k] = x[(size_t)(c0 + k) * NLB_N + n0 + n];
        }
        __syncthreads();

        HV a;
        const float* wrow = W + (size_t)(o0 + r) * NLB_C + c0 + g * 8;
        #pragma unroll
        for (int i = 0; i < 8; i++) { a.h[i]     = (_Float16)wrow[i]; }
        #pragma unroll
        for (int i = 0; i < 8; i++) { a.h[8 + i] = (_Float16)wrow[16 + i]; }

        HV b;
        const float* xcol = ldsXT + r * 32 + g * 8;
        #pragma unroll
        for (int i = 0; i < 16; i++) b.h[i] = (_Float16)xcol[i + (i >> 3) * 8];

        acc = wmma_f16(a.v, b.v, acc);
        __syncthreads();
    }

    #pragma unroll
    for (int i = 0; i < 8; i++) {
        int o = o0 + g * 8 + i;
        float val = acc[i] + bias[o];
        dst[(size_t)(n0 + r) * NLB_CR + o] = (_Float16)val;
    }
}

// ---------------------------------------------------------------------------
// Stage 2: flash attention, TDM-double-buffered K tiles.
// grid = N/64, block = 128 (4 waves, 16 query rows per wave)
// ---------------------------------------------------------------------------
__global__ __launch_bounds__(128) void nlb_attn_kernel(
    const _Float16* __restrict__ Qh,
    const _Float16* __restrict__ Kh,
    const _Float16* __restrict__ Vh,
    _Float16* __restrict__ Oh)
{
    __shared__ __attribute__((aligned(32))) _Float16 ldsK [2][64 * 128]; // ping-pong [key][c]
    __shared__ __attribute__((aligned(32))) _Float16 ldsVt[128 * 64];    // [c][key]
    __shared__ __attribute__((aligned(32))) _Float16 ldsP [4 * 16 * 64];

    const int wave = threadIdx.x >> 5;
    const int lane = threadIdx.x & 31;
    const int g    = lane >> 4;
    const int r    = lane & 15;
    const int q0   = blockIdx.x * 64 + wave * 16;

    // Q A-fragments: 4 head-dim chunks of 32
    HV qa[4];
    #pragma unroll
    for (int c = 0; c < 4; c++) {
        const _Float16* p = Qh + (size_t)(q0 + r) * NLB_CR + c * 32 + g * 8;
        qa[c].h2[0] = *(const v8h*)p;
        qa[c].h2[1] = *(const v8h*)(p + 16);
    }

    v8f oacc[8];
    #pragma unroll
    for (int t = 0; t < 8; t++) oacc[t] = (v8f){};
    float rmax[8], rsum[8];
    #pragma unroll
    for (int i = 0; i < 8; i++) { rmax[i] = -1e30f; rsum[i] = 0.f; }

    _Float16* pw = ldsP + wave * 16 * 64;
    const unsigned ldsK_base[2] = {
        (unsigned)(unsigned long long)(const void*)&ldsK[0][0],
        (unsigned)(unsigned long long)(const void*)&ldsK[1][0]
    };

    // preload K tile 0 via TDM (wave 0 only; EXEC-independent DMA)
    if (wave == 0) tdm_load_k_tile(Kh, ldsK_base[0]);

    for (int kt = 0; kt < NLB_NT; ++kt) {
        const int key0 = kt * 64;
        const int buf  = kt & 1;

        // prefetch next K tile into the other buffer
        if (wave == 0 && kt + 1 < NLB_NT)
            tdm_load_k_tile(Kh + (size_t)(key0 + 64) * NLB_CR, ldsK_base[buf ^ 1]);

        // cooperative transposed staging of V tile (global -> VGPR -> LDS)
        #pragma unroll 2
        for (int j = threadIdx.x; j < 1024; j += 128) {
            int key = j >> 4;
            int c8  = (j & 15) * 8;
            v8h vv = *(const v8h*)(Vh + (size_t)(key0 + key) * NLB_CR + c8);
            #pragma unroll
            for (int w = 0; w < 8; w++) ldsVt[(c8 + w) * 64 + key] = vv[w];
        }

        // current K tile must have landed (TENSORcnt is in-order per wave)
        if (wave == 0) {
            if (kt + 1 < NLB_NT) __builtin_amdgcn_s_wait_tensorcnt(1);
            else                 __builtin_amdgcn_s_wait_tensorcnt(0);
        }
        __syncthreads();

        const _Float16* kb_lds = &ldsK[buf][0];

        // ---- S = Q . K^T  (4 col tiles x 4 k-chunks = 16 WMMAs) ----
        v8f S[4];
        #pragma unroll
        for (int s = 0; s < 4; s++) {
            v8f z = {};
            #pragma unroll
            for (int c = 0; c < 4; c++) {
                HV kb;
                kb.h2[0] = *(const v8h*)(kb_lds + (s * 16 + r) * 128 + c * 32 + g * 8);
                kb.h2[1] = *(const v8h*)(kb_lds + (s * 16 + r) * 128 + c * 32 + g * 8 + 16);
                z = wmma_f16(qa[c].v, kb.v, z);
            }
            S[s] = z * NLB_SCALE;
        }

        // ---- online softmax (row = 8*g + i, reduce over 16-lane half) ----
        float nmax[8], alpha[8];
        #pragma unroll
        for (int i = 0; i < 8; i++) {
            float m = fmaxf(fmaxf(S[0][i], S[1][i]), fmaxf(S[2][i], S[3][i]));
            m = fmaxf(m, __shfl_xor(m, 1, 32));
            m = fmaxf(m, __shfl_xor(m, 2, 32));
            m = fmaxf(m, __shfl_xor(m, 4, 32));
            m = fmaxf(m, __shfl_xor(m, 8, 32));
            nmax[i]  = fmaxf(rmax[i], m);
            alpha[i] = __expf(rmax[i] - nmax[i]);
        }
        #pragma unroll
        for (int i = 0; i < 8; i++) {
            float rs = 0.f;
            #pragma unroll
            for (int s = 0; s < 4; s++) {
                float p = __expf(S[s][i] - nmax[i]);
                S[s][i] = p;
                rs += p;
            }
            rs += __shfl_xor(rs, 1, 32);
            rs += __shfl_xor(rs, 2, 32);
            rs += __shfl_xor(rs, 4, 32);
            rs += __shfl_xor(rs, 8, 32);
            rsum[i] = rsum[i] * alpha[i] + rs;
            rmax[i] = nmax[i];
        }

        // ---- D-layout -> A-layout reshape of P through per-wave LDS ----
        #pragma unroll
        for (int s = 0; s < 4; s++)
            #pragma unroll
            for (int i = 0; i < 8; i++)
                pw[(g * 8 + i) * 64 + s * 16 + r] = (_Float16)S[s][i];
        asm volatile("s_wait_dscnt 0" ::: "memory");

        HV pa[2];
        #pragma unroll
        for (int cc = 0; cc < 2; cc++) {
            const _Float16* pp = pw + r * 64 + cc * 32 + g * 8;
            pa[cc].h2[0] = *(const v8h*)pp;
            pa[cc].h2[1] = *(const v8h*)(pp + 16);
        }

        // ---- O = alpha*O + P.V  (8 col tiles x 2 k-chunks = 16 WMMAs) ----
        #pragma unroll
        for (int t = 0; t < 8; t++) {
            v8f o = oacc[t];
            #pragma unroll
            for (int i = 0; i < 8; i++) o[i] *= alpha[i];
            #pragma unroll
            for (int cc = 0; cc < 2; cc++) {
                HV vb;
                vb.h2[0] = *(const v8h*)(ldsVt + (t * 16 + r) * 64 + cc * 32 + g * 8);
                vb.h2[1] = *(const v8h*)(ldsVt + (t * 16 + r) * 64 + cc * 32 + g * 8 + 16);
                o = wmma_f16(pa[cc].v, vb.v, o);
            }
            oacc[t] = o;
        }
        __syncthreads();
    }

    // ---- normalize and store f16 O[n][c] ----
    #pragma unroll
    for (int t = 0; t < 8; t++)
        #pragma unroll
        for (int i = 0; i < 8; i++) {
            float v = oacc[t][i] / rsum[i];
            Oh[(size_t)(q0 + g * 8 + i) * NLB_CR + t * 16 + r] = (_Float16)v;
        }
}

// ---------------------------------------------------------------------------
// Stage 3: out[o][n] = Wo[o][:] . O[n][:] + bo[o] + x[o][n]   (f32)
// grid = (N/16, 2), block = 256 (8 waves; o0 = (by*8+wave)*16)
// ---------------------------------------------------------------------------
__global__ __launch_bounds__(256) void nlb_outproj_kernel(
    const float* __restrict__ x,
    const float* __restrict__ Wo, const float* __restrict__ bo,
    const _Float16* __restrict__ Oh,
    float* __restrict__ out)
{
    const int wave = threadIdx.x >> 5;
    const int lane = threadIdx.x & 31;
    const int g    = lane >> 4;
    const int r    = lane & 15;
    const int n0   = blockIdx.x * 16;
    const int o0   = (blockIdx.y * 8 + wave) * 16;

    v8f acc = {};
    #pragma unroll
    for (int c0 = 0; c0 < NLB_CR; c0 += 32) {
        HV a, b;
        const float* wrow = Wo + (size_t)(o0 + r) * NLB_CR + c0 + g * 8;
        #pragma unroll
        for (int i = 0; i < 8; i++) { a.h[i]     = (_Float16)wrow[i]; }
        #pragma unroll
        for (int i = 0; i < 8; i++) { a.h[8 + i] = (_Float16)wrow[16 + i]; }
        b.h2[0] = *(const v8h*)(Oh + (size_t)(n0 + r) * NLB_CR + c0 + g * 8);
        b.h2[1] = *(const v8h*)(Oh + (size_t)(n0 + r) * NLB_CR + c0 + g * 8 + 16);
        acc = wmma_f16(a.v, b.v, acc);
    }

    #pragma unroll
    for (int i = 0; i < 8; i++) {
        int o = o0 + g * 8 + i;
        int n = n0 + r;
        out[(size_t)o * NLB_N + n] = acc[i] + bo[o] + x[(size_t)o * NLB_N + n];
    }
}

// ---------------------------------------------------------------------------
extern "C" void kernel_launch(void* const* d_in, const int* in_sizes, int n_in,
                              void* d_out, int out_size, void* d_ws, size_t ws_size,
                              hipStream_t stream) {
    (void)in_sizes; (void)n_in; (void)out_size; (void)ws_size;

    const float* x  = (const float*)d_in[0];
    const float* Wq = (const float*)d_in[1];
    const float* bq = (const float*)d_in[2];
    const float* Wk = (const float*)d_in[3];
    const float* bk = (const float*)d_in[4];
    const float* Wv = (const float*)d_in[5];
    const float* bv = (const float*)d_in[6];
    const float* Wo = (const float*)d_in[7];
    const float* bo = (const float*)d_in[8];
    float* out = (float*)d_out;

    // workspace layout: Qh | Kh | Vh | Oh, each N*CR f16 (4 MB) -> 16 MB total
    const size_t frame = (size_t)NLB_N * NLB_CR;
    _Float16* Qh = (_Float16*)d_ws;
    _Float16* Kh = Qh + frame;
    _Float16* Vh = Kh + frame;
    _Float16* Oh = Vh + frame;

    nlb_qkv_kernel<<<dim3(NLB_N / 16, 3), 256, 0, stream>>>(
        x, Wq, bq, Wk, bk, Wv, bv, Qh, Kh, Vh);

    nlb_attn_kernel<<<dim3(NLB_N / 64), 128, 0, stream>>>(Qh, Kh, Vh, Oh);

    nlb_outproj_kernel<<<dim3(NLB_N / 16, 2), 256, 0, stream>>>(
        x, Wo, bo, Oh, out);
}
